// GraphConvolution_26620207300625
// MI455X (gfx1250) — matrix operand order repaired and checked by reading, not compiled, not run on hardware
//
#include <hip/hip_runtime.h>
#include <hip/hip_bf16.h>

typedef float    v2f  __attribute__((ext_vector_type(2)));
typedef float    v8f  __attribute__((ext_vector_type(8)));
typedef _Float16 v16h __attribute__((ext_vector_type(16)));

#define DIM 128   // feature dim (reference D)

// ---------------------------------------------------------------------------
// Kernel A: segmented aggregation + degree normalization.
// edge_row = repeat(arange(N), deg) => CSR with fixed degree, no atomics.
// One wave (32 lanes) per node; lane l owns feature chunk [4l, 4l+4).
// Edge (col,val) pairs loaded once into lanes 0..deg-1, broadcast via shfl.
// ---------------------------------------------------------------------------
__global__ void gcn_aggregate(const int*   __restrict__ ecol,
                              const float* __restrict__ evalv,
                              const float* __restrict__ feats,
                              float*       __restrict__ hout,
                              int nNodes, int deg)
{
    const int lane = threadIdx.x & 31;
    const int node = (int)((blockIdx.x * blockDim.x + threadIdx.x) >> 5);
    if (node >= nNodes) return;

    const long base = (long)node * deg;
    int   col = 0;
    float val = 0.0f;
    if (lane < deg) {            // deg == 16 here (deg <= 32 assumed)
        col = ecol[base + lane];
        val = evalv[base + lane];
    }

    float4 acc = make_float4(0.f, 0.f, 0.f, 0.f);
    float  dsum = 0.0f;
    for (int e = 0; e < deg; ++e) {
        const int   c = __shfl(col, e, 32);
        const float v = __shfl(val, e, 32);
        const float4 f = ((const float4*)(feats + (long)c * DIM))[lane];
        acc.x += v * f.x; acc.y += v * f.y;
        acc.z += v * f.z; acc.w += v * f.w;
        dsum  += v;
    }
    const float inv = 1.0f / dsum;   // every node has deg>0 edges
    acc.x *= inv; acc.y *= inv; acc.z *= inv; acc.w *= inv;
    ((float4*)(hout + (long)node * DIM))[lane] = acc;
}

// ---------------------------------------------------------------------------
// Kernel B: out = feats + relu(h @ W^T + b), tiled WMMA GEMM.
// Block = 256 threads (8 waves) handles a 16-row M-tile x all 128 columns.
// Wave w owns N-tile [16w, 16w+16). A-tile staged in LDS (16x128 fp32, 8KB).
// C[i][j] = sum_k h[i][k] * W[j][k]  => B-matrix element B[k][n] = W[n][k].
// ---------------------------------------------------------------------------
__global__ void gcn_gemm(const float* __restrict__ h,
                         const float* __restrict__ W,
                         const float* __restrict__ bias,
                         const float* __restrict__ feats,
                         float*       __restrict__ out,
                         int nNodes)
{
    __shared__ __align__(16) float Atile[16 * DIM];

    const long row0 = (long)blockIdx.x * 16;
    const int  t    = threadIdx.x;

    // Cooperative A-tile load: 2048 floats = 512 float4; 2 per thread.
    {
        float4* dst = (float4*)Atile;
        const float4* src = (const float4*)(h + row0 * DIM);
        #pragma unroll
        for (int i = 0; i < 2; ++i) {
            const int idx = t + i * 256;       // [0, 512)
            const int r   = idx >> 5;          // 32 float4 per row
            float4 v = make_float4(0.f, 0.f, 0.f, 0.f);
            if (row0 + r < nNodes) v = src[idx];
            dst[idx] = v;
        }
    }
    __syncthreads();

    const int wave  = t >> 5;
    const int lane  = t & 31;
    const int m     = lane & 15;          // A row / output col-in-tile
    const int n     = wave * 16 + m;      // global output column (W row)
    const int half  = (lane < 16) ? 0 : 1;

    v8f c = {};

#if __has_builtin(__builtin_amdgcn_wmma_f32_16x16x4_f32)
    // Full-precision fp32 tensor path: V_WMMA_F32_16X16X4_F32.
    // A 16x4: lanes 0-15 hold K = k,k+1 ; lanes 16-31 hold K = k+2,k+3.
    // B 4x16: same K split across lane halves, N = lane%16.
    const int koff = half * 2;
    for (int k = 0; k < DIM; k += 4) {
        v2f a = *(const v2f*)&Atile[m * DIM + k + koff];
        v2f b = *(const v2f*)(W + (long)n * DIM + k + koff);
        c = __builtin_amdgcn_wmma_f32_16x16x4_f32(
                false, a, false, b, (short)0, c, false, false);
    }
#else
    // Fallback: codegen-confirmed V_WMMA_F32_16X16X32_F16 (cvt fp32->fp16).
    // A 16x32 f16: lanes 0-15 halves = K {k..k+7, k+16..k+23};
    //              lanes 16-31 halves = K {k+8..k+15, k+24..k+31}.
    // B 32x16 f16: same K split as A.
    const int koffA = half * 8;
    const int koffB = half * 8;
    for (int k = 0; k < DIM; k += 32) {
        v16h a, b;
        #pragma unroll
        for (int i = 0; i < 8; ++i) {
            a[i]     = (_Float16)Atile[m * DIM + k + koffA + i];
            a[i + 8] = (_Float16)Atile[m * DIM + k + 16 + koffA + i];
        }
        const float* wp = W + (long)n * DIM + k + koffB;
        #pragma unroll
        for (int i = 0; i < 8; ++i) { b[i] = (_Float16)wp[i]; b[i + 8] = (_Float16)wp[16 + i]; }
        c = __builtin_amdgcn_wmma_f32_16x16x32_f16(
                false, a, false, b, (short)0, c, false, false);
    }
#endif

    // Epilogue: bias + relu + residual. C/D layout: VGPR v -> row v (+8 for
    // upper lane half), col = lane%16.
    const float bn = bias[n];
    #pragma unroll
    for (int v = 0; v < 8; ++v) {
        const int  mrow = v + half * 8;
        const long r    = row0 + mrow;
        if (r < nNodes) {
            const long idx = r * DIM + n;
            float x = c[v] + bn;
            x = x > 0.0f ? x : 0.0f;
            out[idx] = feats[idx] + x;
        }
    }
}

// ---------------------------------------------------------------------------
// Inputs (setup_inputs order): 0=edge_row 1=edge_col 2=edge_val 3=feats 4=W 5=b
// ---------------------------------------------------------------------------
extern "C" void kernel_launch(void* const* d_in, const int* in_sizes, int n_in,
                              void* d_out, int out_size, void* d_ws, size_t ws_size,
                              hipStream_t stream)
{
    const int*   ecol  = (const int*)  d_in[1];
    const float* evalv = (const float*)d_in[2];
    const float* feats = (const float*)d_in[3];
    const float* W     = (const float*)d_in[4];
    const float* bias  = (const float*)d_in[5];
    float*       out   = (float*)d_out;

    const int E   = in_sizes[1];
    const int N   = in_sizes[3] / DIM;
    const int deg = E / N;                 // 16 (fixed-degree CSR)

    float* hbuf = (float*)d_ws;            // N*DIM fp32 intermediate (51.2 MB)

    // One wave per node: N*32 threads, 256/block.
    const int blocksA = (N * 32 + 255) / 256;
    gcn_aggregate<<<blocksA, 256, 0, stream>>>(ecol, evalv, feats, hbuf, N, deg);

    // One block per 16-row tile.
    const int tiles = (N + 15) / 16;
    gcn_gemm<<<tiles, 256, 0, stream>>>(hbuf, W, bias, feats, out, N);
}
